// MultiHeadSelfAttention_17892833755338
// MI455X (gfx1250) — matrix operand order repaired and checked by reading, not compiled, not run on hardware
//
#include <hip/hip_runtime.h>
#include <hip/hip_bf16.h>
#include <cstddef>
#include <cstdint>

// ---------------------------------------------------------------------------
// Types for CDNA5 WMMA (wave32): V_WMMA_F32_16X16X32_BF16
// ---------------------------------------------------------------------------
typedef __attribute__((ext_vector_type(16))) __bf16 v16bf;
typedef __attribute__((ext_vector_type(8)))  __bf16 v8bf;
typedef __attribute__((ext_vector_type(8)))  float  v8f;

static __device__ inline v8f wmma_bf16(v16bf a, v16bf b, v8f c) {
  return __builtin_amdgcn_wmma_f32_16x16x32_bf16(
      /*neg_a=*/false, a, /*neg_b=*/false, b,
      /*c_mod=*/(short)0, c, /*reuse_a=*/false, /*reuse_b=*/false);
}

static __device__ inline v8bf ld8(const __bf16* p) {
  return *reinterpret_cast<const v8bf*>(p);
}

static __device__ inline v16bf cat16(v8bf lo, v8bf hi) {
  return __builtin_shufflevector(lo, hi, 0,1,2,3,4,5,6,7,8,9,10,11,12,13,14,15);
}

// float -> bf16 round-to-nearest-even
static __device__ inline __bf16 f2bf(float f) {
  unsigned u = __builtin_bit_cast(unsigned, f);
  u = (u + 0x7FFFu + ((u >> 16) & 1u)) >> 16;
  unsigned short s = (unsigned short)u;
  return __builtin_bit_cast(__bf16, s);
}

// ---------------------------------------------------------------------------
// CDNA5 async global->LDS DMA (no VGPR round trip; tracked by ASYNCcnt).
//   LDS-offset VGPR operand is derived from the REAL pointer (ptrtoint) so
//   the shared object escapes into the memory-clobbering asm.
//   16 bytes per lane.  CDNA5 ISA.
// ---------------------------------------------------------------------------
static __device__ inline void async_ld16_to_lds(__bf16* l, const void* g) {
  unsigned ldsoff = (unsigned)(uintptr_t)l;   // low 32 bits = LDS byte offset
  asm volatile("global_load_async_to_lds_b128 %0, %1, off"
               :: "v"(ldsoff), "v"(g)
               : "memory");
}
static __device__ inline void wait_asynccnt0() {
  asm volatile("s_wait_asynccnt 0x0" ::: "memory");
}

// ---------------------------------------------------------------------------
// Problem constants
// ---------------------------------------------------------------------------
#define BATCH   2
#define SEQ     2048
#define EMB     1024
#define HEADS   16
#define HDIM    64
#define MROWS   (BATCH * SEQ)          // 4096
#define PER_BH  (SEQ * HDIM)           // 131072
#define PER_B   (HEADS * PER_BH)       // 2097152

// ---------------------------------------------------------------------------
// Kernel 0: f32 -> bf16 elementwise conversion
// ---------------------------------------------------------------------------
__global__ __launch_bounds__(256) void cvt_bf16_kernel(
    const float* __restrict__ in, __bf16* __restrict__ out, int n) {
  int i = blockIdx.x * blockDim.x + threadIdx.x;
  if (i < n) out[i] = f2bf(in[i]);
}

// ---------------------------------------------------------------------------
// Kernel 1: GEMM  out[m,n] = sum_k A[m,k] * W[n,k] + bias[n]
//   A: [4096,1024] bf16 row-major, W: [1024,1024] bf16 row-major ([out,in]).
//   Block = 8 waves, block tile 128(M) x 64(N); wave tile 16(M) x 64(N).
//   B chunk (32K x 64N, 4 KB) staged in double-buffered LDS, filled one
//   K-step ahead by ASYNC DMA (global_load_async_to_lds_b128); only in-loop
//   fill synchronization is s_wait_asynccnt at the barrier. A operand is
//   register double-buffered; all 4 B operands are pulled from LDS into
//   distinct registers BEFORE the WMMAs so the 8 ds_load_b128s clause under
//   a single s_wait_dscnt and the 4 WMMAs issue back-to-back.
//   Blds is zero-initialized with REAL C++ stores once at entry so the
//   compiler cannot undef-fold the ds_loads (DMA writes are asm-only).
//   mode 0: f32 out[m*1024+n]                     (final projection)
//   mode 1: bf16 out[b,h,s,d]  (Q,K heads split)
//   mode 2: bf16 out[b,h,d,s]  (V transposed for PV WMMA B-operand)
// ---------------------------------------------------------------------------
__global__ __launch_bounds__(256) void gemm_bf16_kernel(
    const __bf16* __restrict__ A, const __bf16* __restrict__ W,
    const float* __restrict__ bias, float* __restrict__ outf,
    __bf16* __restrict__ outb, int mode) {
  // LDS: [buf][col 0..63][k 0..31] bf16; buf stride = 4096 bytes.
  __shared__ __align__(16) __bf16 Blds[2][64 * 32];

  const int tid  = threadIdx.x;
  const int lane = tid & 31;
  const int w    = tid >> 5;
  const int mblk = blockIdx.x >> 4;   // 0..31
  const int nt   = blockIdx.x & 15;   // 0..15
  const int mbase = mblk * 128 + w * 16;
  const int nbase = nt * 64;
  const int row  = lane & 15;
  const int hi   = (lane >> 4) & 1;

  // cooperative fill mapping: thread -> (col, ksub); 16 B per thread
  const int fcol  = tid >> 2;          // 0..63
  const int fksub = (tid & 3) * 8;     // 0,8,16,24
  const int fidx  = fcol * 32 + fksub; // element index inside one buffer
  const __bf16* wfill = W + (size_t)(nbase + fcol) * EMB + fksub;

  const __bf16* arow = A + (size_t)(mbase + row) * EMB;

  v8f zero = {};
  v8f acc[4];
#pragma unroll
  for (int t = 0; t < 4; ++t) acc[t] = zero;

  // real C++ stores into LDS (prevents undef-folding of later ds_loads);
  // barrier orders them before the DMA writes to the same addresses
  v8bf bz = {};
  *reinterpret_cast<v8bf*>(&Blds[0][fidx]) = bz;
  *reinterpret_cast<v8bf*>(&Blds[1][fidx]) = bz;
  __syncthreads();

  // prologue: async-fill buffer 0, preload first A chunk into registers
  async_ld16_to_lds(&Blds[0][fidx], wfill);
  const int ka0 = hi * 8;
  v8bf alo = ld8(arow + ka0);
  v8bf ahi = ld8(arow + ka0 + 16);
  wait_asynccnt0();
  __syncthreads();

  int buf = 0;
  for (int kk = 0; kk < EMB; kk += 32) {
    // async-fill next chunk into the other buffer (overlaps the WMMAs)
    if (kk + 32 < EMB)
      async_ld16_to_lds(&Blds[buf ^ 1][fidx], wfill + kk + 32);

    v16bf a = cat16(alo, ahi);

    // preload next A chunk (consumed next iteration)
    if (kk + 32 < EMB) {
      const int ka = kk + 32 + hi * 8;
      alo = ld8(arow + ka);
      ahi = ld8(arow + ka + 16);
      __builtin_prefetch(arow + ka + 32, 0, 3);
    }

    // pull ALL 4 B operands from LDS first (distinct regs -> one dscnt wait)
    const __bf16* bb = &Blds[buf][(size_t)row * 32 + hi * 8];
    v16bf b4[4];
#pragma unroll
    for (int t = 0; t < 4; ++t) {
      const __bf16* bp = bb + t * (16 * 32);
      b4[t] = cat16(ld8(bp), ld8(bp + 16));
    }
#pragma unroll
    for (int t = 0; t < 4; ++t)
      acc[t] = wmma_bf16(a, b4[t], acc[t]);

    wait_asynccnt0();   // our async fill of buf^1 is done
    __syncthreads();    // everyone's fill of buf^1 done / reads of buf done
    buf ^= 1;
  }

  // D layout: vgpr r -> row m = r + hi*8 ; N = lane%16
#pragma unroll
  for (int t = 0; t < 4; ++t) {
    const int ncol = nbase + t * 16 + row;
    const float bv = bias[ncol];
#pragma unroll
    for (int r = 0; r < 8; ++r) {
      const int m = mbase + r + hi * 8;
      const float val = acc[t][r] + bv;
      if (mode == 0) {
        outf[(size_t)m * EMB + ncol] = val;
      } else {
        const int b = m >> 11;        // / SEQ
        const int s = m & (SEQ - 1);
        const int h = ncol >> 6;      // / HDIM
        const int d = ncol & (HDIM - 1);
        size_t idx;
        if (mode == 1)
          idx = (size_t)b * PER_B + (size_t)h * PER_BH + (size_t)s * HDIM + d;
        else
          idx = (size_t)b * PER_B + (size_t)h * PER_BH + (size_t)d * SEQ + s;
        outb[idx] = f2bf(val);
      }
    }
  }
}

// ---------------------------------------------------------------------------
// Kernel 2: flash attention. One wave per (b, h, 16-query tile).
//   Q,K: bf16 [B,H,S,Dh] ; Vt: bf16 [B,H,Dh,S] ; C: bf16 [B,S,E] concat.
//   Streams keys in chunks of 32, online softmax, never materializes scores.
//   All K and V loads for a chunk issue up front so the V fetch hides
//   behind the softmax VALU work (partial loadcnt waits).
// ---------------------------------------------------------------------------
__global__ __launch_bounds__(256) void attn_kernel(
    const __bf16* __restrict__ Q, const __bf16* __restrict__ K,
    const __bf16* __restrict__ Vt, __bf16* __restrict__ C) {
  __shared__ __align__(16) __bf16 Plds[8][16 * 32];  // per-wave P strip

  const int lane = threadIdx.x & 31;
  const int w    = threadIdx.x >> 5;
  const int wid  = blockIdx.x * 8 + w;       // 0..4095
  const int bh   = wid >> 7;                 // 0..31
  const int qt   = wid & 127;                // query tile
  const int b    = bh >> 4, h = bh & 15;
  const size_t base = (size_t)bh * PER_BH;
  const int row = lane & 15;
  const int hi  = (lane >> 4) & 1;
  __bf16* P = Plds[w];

  // Q as A-operand (rows = queries, K-dim = head dim). Held for whole loop.
  const __bf16* qp = Q + base + (size_t)(qt * 16 + row) * HDIM + hi * 16;
  v16bf aQ0 = cat16(ld8(qp),      ld8(qp + 8));    // d = 0..31
  v16bf aQ1 = cat16(ld8(qp + 32), ld8(qp + 40));   // d = 32..63

  v8f zero = {};
  v8f acc[4];
  float mI[8], lI[8];
#pragma unroll
  for (int t = 0; t < 4; ++t) acc[t] = zero;
#pragma unroll
  for (int r = 0; r < 8; ++r) { mI[r] = -1e30f; lI[r] = 0.f; }

  for (int j = 0; j < SEQ; j += 32) {
    // ---- issue ALL loads for this chunk up front ----
    const __bf16* kp0 = K + base + (size_t)(j + row) * HDIM + hi * 16;
    const __bf16* kp1 = kp0 + 16 * HDIM;
    v16bf kb00 = cat16(ld8(kp0),      ld8(kp0 + 8));
    v16bf kb01 = cat16(ld8(kp0 + 32), ld8(kp0 + 40));
    v16bf kb10 = cat16(ld8(kp1),      ld8(kp1 + 8));
    v16bf kb11 = cat16(ld8(kp1 + 32), ld8(kp1 + 40));
    v16bf vb[4];
#pragma unroll
    for (int t = 0; t < 4; ++t) {
      const __bf16* vp = Vt + base + (size_t)(t * 16 + row) * SEQ + j + hi * 16;
      vb[t] = cat16(ld8(vp), ld8(vp + 8));
    }

    // scores: two 16x16 key tiles, contraction over d (two WMMAs each)
    v8f s0 = zero, s1 = zero;
    s0 = wmma_bf16(aQ0, kb00, s0);
    s0 = wmma_bf16(aQ1, kb01, s0);
    s1 = wmma_bf16(aQ0, kb10, s1);
    s1 = wmma_bf16(aQ1, kb11, s1);

    // online softmax per row (D layout: vgpr r -> row r + hi*8, N = lane%16)
#pragma unroll
    for (int r = 0; r < 8; ++r) {
      float a0 = s0[r] * 0.125f;   // 1/sqrt(64)
      float a1 = s1[r] * 0.125f;
      float mx = fmaxf(a0, a1);
      mx = fmaxf(mx, __shfl_xor(mx, 1));
      mx = fmaxf(mx, __shfl_xor(mx, 2));
      mx = fmaxf(mx, __shfl_xor(mx, 4));
      mx = fmaxf(mx, __shfl_xor(mx, 8));   // stays inside 16-lane half
      float mnew = fmaxf(mI[r], mx);
      float al = __expf(mI[r] - mnew);
      float p0 = __expf(a0 - mnew);
      float p1 = __expf(a1 - mnew);
      float rs = p0 + p1;
      rs += __shfl_xor(rs, 1);
      rs += __shfl_xor(rs, 2);
      rs += __shfl_xor(rs, 4);
      rs += __shfl_xor(rs, 8);
      lI[r] = lI[r] * al + rs;
      mI[r] = mnew;
      acc[0][r] *= al; acc[1][r] *= al; acc[2][r] *= al; acc[3][r] *= al;
      // stage P (D layout -> row-major [16 q][32 keys]) for A-layout reload
      P[(r + hi * 8) * 32 + row]      = f2bf(p0);
      P[(r + hi * 8) * 32 + 16 + row] = f2bf(p1);
    }

    // Same-wave LDS ops are in-order (DScnt): loads below observe stores above.
    const __bf16* pp = P + row * 32 + hi * 16;
    v16bf pA = cat16(ld8(pp), ld8(pp + 8));    // A: 16 q x 32 keys

    // PV: V operands already in registers (loaded at chunk top)
#pragma unroll
    for (int t = 0; t < 4; ++t)
      acc[t] = wmma_bf16(pA, vb[t], acc[t]);
  }

  // normalize and write concat[b, s, h*64 + d] (bf16)
#pragma unroll
  for (int r = 0; r < 8; ++r) {
    const float inv = 1.0f / lI[r];
    const int s = qt * 16 + r + hi * 8;
#pragma unroll
    for (int t = 0; t < 4; ++t) {
      const int d = t * 16 + row;
      C[(size_t)(b * SEQ + s) * EMB + h * HDIM + d] = f2bf(acc[t][r] * inv);
    }
  }
}

// ---------------------------------------------------------------------------
// Launcher
// ---------------------------------------------------------------------------
extern "C" void kernel_launch(void* const* d_in, const int* in_sizes, int n_in,
                              void* d_out, int out_size, void* d_ws, size_t ws_size,
                              hipStream_t stream) {
  (void)in_sizes; (void)n_in; (void)out_size; (void)ws_size;

  const float* X  = (const float*)d_in[0];
  const float* Wq = (const float*)d_in[1];
  const float* bq = (const float*)d_in[2];
  const float* Wk = (const float*)d_in[3];
  const float* bk = (const float*)d_in[4];
  const float* Wv = (const float*)d_in[5];
  const float* bv = (const float*)d_in[6];
  const float* Wo = (const float*)d_in[7];
  const float* bo = (const float*)d_in[8];
  float* out = (float*)d_out;

  // workspace layout (bytes)
  char* ws = (char*)d_ws;
  const size_t SZ_X  = (size_t)MROWS * EMB * 2;   // 8 MB
  const size_t SZ_W  = (size_t)EMB * EMB * 2;     // 2 MB
  __bf16* Xb  = (__bf16*)(ws);
  __bf16* Wqb = (__bf16*)(ws + SZ_X);
  __bf16* Wkb = (__bf16*)(ws + SZ_X + SZ_W);
  __bf16* Wvb = (__bf16*)(ws + SZ_X + 2 * SZ_W);
  __bf16* Wob = (__bf16*)(ws + SZ_X + 3 * SZ_W);
  __bf16* Qb  = (__bf16*)(ws + SZ_X + 4 * SZ_W);
  __bf16* Kb  = (__bf16*)(ws + 2 * SZ_X + 4 * SZ_W);
  __bf16* Vtb = (__bf16*)(ws + 3 * SZ_X + 4 * SZ_W);
  __bf16* Cb  = (__bf16*)(ws + 4 * SZ_X + 4 * SZ_W);

  // 0) convert activations + weights to bf16
  {
    int nX = MROWS * EMB, nW = EMB * EMB;
    cvt_bf16_kernel<<<(nX + 255) / 256, 256, 0, stream>>>(X,  Xb,  nX);
    cvt_bf16_kernel<<<(nW + 255) / 256, 256, 0, stream>>>(Wq, Wqb, nW);
    cvt_bf16_kernel<<<(nW + 255) / 256, 256, 0, stream>>>(Wk, Wkb, nW);
    cvt_bf16_kernel<<<(nW + 255) / 256, 256, 0, stream>>>(Wv, Wvb, nW);
    cvt_bf16_kernel<<<(nW + 255) / 256, 256, 0, stream>>>(Wo, Wob, nW);
  }

  // 1) QKV projections: 32 Mblk x 16 Nblk = 512 blocks of 8 waves
  gemm_bf16_kernel<<<512, 256, 0, stream>>>(Xb, Wqb, bq, nullptr, Qb, 1);
  gemm_bf16_kernel<<<512, 256, 0, stream>>>(Xb, Wkb, bk, nullptr, Kb, 1);
  gemm_bf16_kernel<<<512, 256, 0, stream>>>(Xb, Wvb, bv, nullptr, Vtb, 2);

  // 2) flash attention: 2*16*128 = 4096 waves -> 512 blocks
  attn_kernel<<<512, 256, 0, stream>>>(Qb, Kb, Vtb, Cb);

  // 3) output projection (fp32 result + bias)
  gemm_bf16_kernel<<<512, 256, 0, stream>>>(Cb, Wob, bo, out, nullptr, 0);
}
